// LSTM_Layer_28698971472165
// MI455X (gfx1250) — matrix operand order repaired and checked by reading, not compiled, not run on hardware
//
#include <hip/hip_runtime.h>

// ---------------- problem sizes ----------------
#define TB 512
#define DB 512
#define UB 512
#define BB 64

// ---------------- CDNA5 types ----------------
typedef __attribute__((ext_vector_type(16))) __bf16       v16bf;
typedef __attribute__((ext_vector_type(8)))  float        v8f;
typedef __attribute__((ext_vector_type(8)))  unsigned int v8u;
typedef int v4i __attribute__((vector_size(16)));          // matches builtin param type

struct U8x32 { uint4 lo; uint4 hi; };   // 32 bytes, bit-cast to v16bf

// ---------------- optional async-to-LDS path (guarded) ----------------
#if defined(__has_builtin)
#if __has_builtin(__builtin_amdgcn_global_load_async_to_lds_b128) && \
    __has_builtin(__builtin_amdgcn_s_wait_asynccnt)
#define LSTM_ASYNC 1
#endif
#endif
#ifndef LSTM_ASYNC
#define LSTM_ASYNC 0
#endif

#if LSTM_ASYNC
typedef __attribute__((address_space(1))) v4i* g_v4i_p;    // global (AS1)
typedef __attribute__((address_space(3))) v4i* l_v4i_p;    // LDS (AS3)
#endif

__device__ __forceinline__ void g2lds_b128(const void* g, void* l) {
#if LSTM_ASYNC
  __builtin_amdgcn_global_load_async_to_lds_b128(
      (g_v4i_p)(unsigned long long)g, (l_v4i_p)l, 0, 0);
#else
  *(uint4*)l = *(const uint4*)g;
#endif
}
__device__ __forceinline__ void wait_async_all() {
#if LSTM_ASYNC
  __builtin_amdgcn_s_wait_asynccnt(0);
#endif
}

// ---------------- helpers ----------------
__device__ __forceinline__ unsigned short f32_to_bf16_bits(float f) {
  unsigned u = __float_as_uint(f);
  u += 0x7FFFu + ((u >> 16) & 1u);          // round-to-nearest-even
  return (unsigned short)(u >> 16);
}
__device__ __forceinline__ float sigmoidf_(float x) {
  return 1.0f / (1.0f + __expf(-x));
}
__device__ __forceinline__ float tanhf_(float x) {
  float e = __expf(-2.0f * x);
  return (1.0f - e) / (1.0f + e);
}

__device__ __forceinline__ void grid_barrier(unsigned int* cnt, unsigned int target) {
  __syncthreads();
  if (threadIdx.x == 0) {
    __hip_atomic_fetch_add(cnt, 1u, __ATOMIC_ACQ_REL, __HIP_MEMORY_SCOPE_AGENT);
    while (__hip_atomic_load(cnt, __ATOMIC_ACQUIRE, __HIP_MEMORY_SCOPE_AGENT) < target) {
      __builtin_amdgcn_s_sleep(2);
    }
  }
  __syncthreads();
}

// ---------------- LDS budget (dynamic smem; 305,408 B < 320 KB/WGP) ----------------
#define ZS          1032                      // z row stride in shorts (16B pad -> bank stagger)
#define WLDS_BYTES  262144                    // 1024K x 128N bf16 weight slice
#define Z_BYTES     (16 * ZS * 2)             // 33,024
#define GACC_BYTES  (4 * 16 * 32 * 4)         // 8,192  pre-activations f/i/g/o
#define C_BYTES     (16 * 32 * 4)             // 2,048  cell state
#define SMEM_BYTES  (WLDS_BYTES + Z_BYTES + GACC_BYTES + C_BYTES)
#define HBUF_ELEMS  (BB * UB)                 // one h ping buffer (bf16 elements)

// ================= kernel 1: pack weights fp32 -> bf16 WMMA-B fragment layout ==========
// Packed index P = ub*131072 + kt*4096 + ntb*512 + lane*16 + e
//   lane<16 : K = kt*32 + e,      lane>=16: K = kt*32 + 16 + e
// gate = ntb>>1 (f,i,g,o), unit u = ub*32 + (ntb&1)*16 + (lane&15)
extern "C" __global__ void lstm_pack_kernel(const float* __restrict__ Wf,
                                            const float* __restrict__ Wi,
                                            const float* __restrict__ Wc,
                                            const float* __restrict__ Wo,
                                            unsigned short* __restrict__ Wp) {
  unsigned P = blockIdx.x * 256u + threadIdx.x;
  if (P >= (1u << 21)) return;                // 2M packed bf16 elements
  int e    = P & 15;
  int lane = (P >> 4) & 31;
  int ntb  = (P >> 9) & 7;
  int kt   = (P >> 12) & 31;
  int ub   = P >> 17;
  int gate = ntb >> 1;
  int u    = ub * 32 + (ntb & 1) * 16 + (lane & 15);
  int k    = kt * 32 + ((lane < 16) ? e : 16 + e);
  const float* W = (gate == 0) ? Wf : (gate == 1) ? Wi : (gate == 2) ? Wc : Wo;
  Wp[P] = f32_to_bf16_bits(W[k * UB + u]);
}

// ================= kernel 2: reset device barrier counter =================
extern "C" __global__ void lstm_init_kernel(unsigned int* cnt) {
  if (threadIdx.x == 0 && blockIdx.x == 0) *cnt = 0u;
}

// ================= kernel 3: fused recurrence =================
// grid = 64 blocks: (mb 0..3) x (ub 0..15). block = 128 threads = 4 waves, wave w = gate w.
// Per step per wave: [16 x 1024] x [1024 x 32] via 64 x v_wmma_f32_16x16x32_bf16.
extern "C" __global__ void __launch_bounds__(128, 1)
lstm_recurrence_kernel(const float* __restrict__ x,
                       const unsigned short* __restrict__ Wp,
                       const float* __restrict__ bfp, const float* __restrict__ bip,
                       const float* __restrict__ bcp, const float* __restrict__ bop,
                       unsigned short* __restrict__ hbuf,      // 2 x [64 x 512] bf16 ping-pong
                       unsigned int* __restrict__ barrier_cnt,
                       float* __restrict__ out)                // [B, T, U] fp32
{
  extern __shared__ char smem[];
  unsigned short* Wlds = (unsigned short*)smem;
  unsigned short* zs   = (unsigned short*)(smem + WLDS_BYTES);
  float* gacc          = (float*)(smem + WLDS_BYTES + Z_BYTES);
  float* cst           = (float*)(smem + WLDS_BYTES + Z_BYTES + GACC_BYTES);

  const int tid   = threadIdx.x;
  const int mb    = blockIdx.x >> 4;
  const int ubk   = blockIdx.x & 15;
  const int m0    = mb * 16;
  const int u0    = ubk * 32;
  const int w     = tid >> 5;       // wave index == gate index (f,i,g,o)
  const int lane  = tid & 31;
  const int l15   = lane & 15;
  const int lhalf = lane >> 4;

  // ---- stage this block's 256 KB weight slice into LDS (once) ----
  {
    const unsigned short* wsrc = Wp + (size_t)ubk * 131072;
    for (int i = tid; i < 16384; i += 128)
      g2lds_b128(wsrc + i * 8, Wlds + i * 8);
  }

  // ---- init cell state + our slice of h0 = 0 ----
  for (int idx = tid; idx < 512; idx += 128) cst[idx] = 0.0f;
  for (int idx = tid; idx < 512; idx += 128) {
    int r = idx >> 5, jj = idx & 31;
    hbuf[(m0 + r) * UB + u0 + jj] = 0;       // hbuf ping 0
  }
  wait_async_all();
  __threadfence();
  grid_barrier(barrier_cnt, 64u);

  // biases hoisted (per-lane, per-gate)
  const float* bias_g = (w == 0) ? bfp : (w == 1) ? bip : (w == 2) ? bcp : bop;
  const float bv0 = bias_g[u0 + l15];
  const float bv1 = bias_g[u0 + 16 + l15];

  const unsigned short* wb0  = Wlds + (w * 2) * 512 + lane * 16;   // N-tile 0 of this gate
  const unsigned short* wb1  = wb0 + 512;                          // N-tile 1
  const unsigned short* arow = zs + l15 * ZS + lhalf * 16;         // A-fragment base

  for (int t = 0; t < TB; ++t) {
    const int cur = t & 1, nxt = cur ^ 1;
    const unsigned short* hsrc = hbuf + cur * HBUF_ELEMS + m0 * UB;

    // ---- build z = [h_t | x_t] in LDS (bf16, row stride ZS) ----
    for (int i = tid; i < 1024; i += 128) {                  // h: 16 rows x 64 b128
      int r = i >> 6, c8 = i & 63;
      g2lds_b128(hsrc + r * UB + c8 * 8, zs + r * ZS + c8 * 8);
    }
    for (int i = tid; i < 2048; i += 128) {                  // x: fp32 -> bf16
      int r = i >> 7, c4 = i & 127;
      float4 v = *(const float4*)(x + (size_t)(m0 + r) * TB * DB + (size_t)t * DB + c4 * 4);
      ushort4 h4;
      h4.x = f32_to_bf16_bits(v.x); h4.y = f32_to_bf16_bits(v.y);
      h4.z = f32_to_bf16_bits(v.z); h4.w = f32_to_bf16_bits(v.w);
      *(ushort4*)(zs + r * ZS + 512 + c4 * 4) = h4;
    }
    if (t + 1 < TB) {                                        // prefetch next x_t
      const float* nx = x + (size_t)(m0 + (tid >> 3)) * TB * DB
                          + (size_t)(t + 1) * DB + (tid & 7) * 64;
      __builtin_prefetch(nx, 0, 1);
    }
    wait_async_all();
    __syncthreads();

    // ---- GEMM: [16 x 1024] @ [1024 x 32] for this wave's gate ----
    v8f acc0 = {};
    v8f acc1 = {};
#pragma unroll 4
    for (int kt = 0; kt < 32; ++kt) {
      const unsigned short* ap = arow + kt * 32;
      U8x32 a8;
      a8.lo = *(const uint4*)(ap);
      a8.hi = *(const uint4*)(ap + 8);
      v16bf av = __builtin_bit_cast(v16bf, a8);
      v16bf b0 = __builtin_bit_cast(v16bf, *(const v8u*)(wb0 + kt * 4096));
      v16bf b1 = __builtin_bit_cast(v16bf, *(const v8u*)(wb1 + kt * 4096));
      acc0 = __builtin_amdgcn_wmma_f32_16x16x32_bf16(false, av, false, b0,
                                                     (short)0, acc0, false, false);
      acc1 = __builtin_amdgcn_wmma_f32_16x16x32_bf16(false, av, false, b1,
                                                     (short)0, acc1, false, false);
    }

    // ---- spill pre-activations (+bias) to LDS ----
#pragma unroll
    for (int p = 0; p < 8; ++p) {
      int M = p + (lhalf << 3);
      gacc[(w * 16 + M) * 32 + l15]      = acc0[p] + bv0;
      gacc[(w * 16 + M) * 32 + 16 + l15] = acc1[p] + bv1;
    }
    __syncthreads();

    // ---- fused gate nonlinearity / state update ----
    unsigned short* hdst = hbuf + nxt * HBUF_ELEMS;
    for (int idx = tid; idx < 512; idx += 128) {
      int r = idx >> 5, jj = idx & 31;
      float fg = sigmoidf_(gacc[(0 * 16 + r) * 32 + jj]);
      float ig = sigmoidf_(gacc[(1 * 16 + r) * 32 + jj]);
      float gg = tanhf_   (gacc[(2 * 16 + r) * 32 + jj]);
      float og = sigmoidf_(gacc[(3 * 16 + r) * 32 + jj]);
      float c  = fg * cst[idx] + ig * gg;
      cst[idx] = c;
      float h  = og * tanhf_(c);
      hdst[(m0 + r) * UB + u0 + jj] = f32_to_bf16_bits(h);
      out[(size_t)(m0 + r) * TB * UB + (size_t)t * UB + u0 + jj] = h;
    }
    __threadfence();
    grid_barrier(barrier_cnt, 64u * (unsigned)(t + 2));
  }
}

// ================= host launcher =================
extern "C" void kernel_launch(void* const* d_in, const int* in_sizes, int n_in,
                              void* d_out, int out_size, void* d_ws, size_t ws_size,
                              hipStream_t stream) {
  (void)in_sizes; (void)n_in; (void)out_size; (void)ws_size;
  const float* x  = (const float*)d_in[0];
  const float* Wf = (const float*)d_in[1];
  const float* bf = (const float*)d_in[2];
  const float* Wi = (const float*)d_in[3];
  const float* bi = (const float*)d_in[4];
  const float* Wc = (const float*)d_in[5];
  const float* bc = (const float*)d_in[6];
  const float* Wo = (const float*)d_in[7];
  const float* bo = (const float*)d_in[8];
  float* out = (float*)d_out;

  // workspace: [0, 4MB) packed bf16 weights | [4MB, +128KB) h ping-pong | +4B barrier
  unsigned short* Wp   = (unsigned short*)d_ws;
  unsigned short* hbuf = (unsigned short*)((char*)d_ws + (size_t)(4u << 20));
  unsigned int*   cnt  = (unsigned int*)((char*)d_ws + (size_t)(4u << 20)
                                         + 2u * HBUF_ELEMS * sizeof(unsigned short));

  (void)hipFuncSetAttribute((const void*)lstm_recurrence_kernel,
                            hipFuncAttributeMaxDynamicSharedMemorySize, SMEM_BYTES);

  lstm_pack_kernel<<<8192, 256, 0, stream>>>(Wf, Wi, Wc, Wo, Wp);
  lstm_init_kernel<<<1, 64, 0, stream>>>(cnt);
  lstm_recurrence_kernel<<<64, 128, SMEM_BYTES, stream>>>(x, Wp, bf, bi, bc, bo,
                                                          hbuf, cnt, out);
}
